// RMoERouter_56710748176492
// MI455X (gfx1250) — matrix operand order repaired and checked by reading, not compiled, not run on hardware
//
#include <hip/hip_runtime.h>
#include <hip/hip_bf16.h>

typedef __attribute__((ext_vector_type(16))) __bf16 v16bf;
typedef __attribute__((ext_vector_type(8)))  __bf16 v8bf;
typedef __attribute__((ext_vector_type(4)))  __bf16 v4bf;
typedef __attribute__((ext_vector_type(8)))  float  v8f;

#define B_   32
#define S_   2048
#define D_   768
#define E_   8
#define H_   256
#define G3_  768      /* 3*H */
#define BS_  65536    /* B*S */
#define NH_  208      /* head N: 128 (c1) + 8 logits + 8 pad + 64 (k1) */

/* ---- output layout (flat float concat in tuple order) ---- */
#define OUT_EW   ((size_t)0)
#define OUT_IDX  ((size_t)262144)
#define OUT_RW   ((size_t)524288)
#define OUT_HID  ((size_t)1048576)
#define OUT_CPLX ((size_t)1056768)
#define OUT_LB   ((size_t)1122304)
#define OUT_CREG ((size_t)1122305)

/* ---- workspace layout (bytes); bf16 weights kept in NATURAL [n][k] order ---- */
#define WS_GI     ((size_t)0)                               /* float[BS*3H]   */
#define WS_G      (WS_GI   + (size_t)BS_*G3_*4)             /* float[BS*H]    */
#define WS_H      (WS_G    + (size_t)BS_*H_*4)              /* float[2][B*H]  */
#define WS_WIH    (WS_H    + (size_t)2*B_*H_*4)             /* bf16[768][768] */
#define WS_WHH    (WS_WIH  + (size_t)G3_*D_*2)              /* bf16[768][256] */
#define WS_WHEAD  (WS_WHH  + (size_t)G3_*H_*2)              /* bf16[208][256] */
#define WS_USAGE  (WS_WHEAD+ (size_t)NH_*H_*2)              /* float[8]       */
#define WS_CSUM   (WS_USAGE + 64)
#define WS_CNT    (WS_CSUM  + 64)

/* ---- Tensor Data Mover path (gfx1250 TDM), guarded for both toolchains ---- */
#if defined(__gfx1250__) && __has_builtin(__builtin_amdgcn_tensor_load_to_lds)
#define USE_TDM 1
#else
#define USE_TDM 0
#endif

#if USE_TDM
typedef unsigned int u32x4 __attribute__((ext_vector_type(4)));
typedef int          i32x4 __attribute__((ext_vector_type(4)));
typedef int          i32x8 __attribute__((ext_vector_type(8)));

/* 2-D bf16 tile DMA: global[tensor_rows][row_stride_elems] -> LDS, tile =
   tile_rows x tile_cols elements. D# packing per ISA 8.3/8.4 (count=1,
   type=2 "image", data_size=2B, no padding/iterate/multicast). */
__device__ inline void tdm_load_2d(void* lds_dst, const void* gsrc,
                                   unsigned tile_rows, unsigned tile_cols,
                                   unsigned tensor_rows, unsigned row_stride_elems) {
  unsigned lds_off = (unsigned)(unsigned long long)lds_dst; /* LDS byte addr */
  unsigned long long ga = (unsigned long long)gsrc;
  u32x4 g0 = { 1u,                                          /* count=1 valid */
               lds_off,                                     /* lds_addr      */
               (unsigned)ga,                                /* gaddr[31:0]   */
               ((unsigned)(ga >> 32) & 0x01FFFFFFu) | (2u << 30) }; /* type=2 */
  i32x8 g1 = { (int)0x00010000,                             /* data_size=2B  */
               (int)((tile_cols & 0xFFFFu) << 16),          /* tensor_dim0 lo */
               (int)(((tile_cols >> 16) & 0xFFFFu) |
                     ((tensor_rows & 0xFFFFu) << 16)),      /* t_dim0hi|t_dim1lo */
               (int)(((tensor_rows >> 16) & 0xFFFFu) |
                     ((tile_cols & 0xFFFFu) << 16)),        /* t_dim1hi|tile_dim0 */
               (int)(tile_rows & 0xFFFFu),                  /* tile_dim1 (dim2=0) */
               (int)row_stride_elems,                       /* dim0_stride lo32   */
               0, 0 };
  i32x4 z4 = {0, 0, 0, 0};
#if __clang_major__ >= 23
  i32x8 z8 = {0, 0, 0, 0, 0, 0, 0, 0};
  __builtin_amdgcn_tensor_load_to_lds(g0, g1, z4, z4, z8, 0);
#else
  __builtin_amdgcn_tensor_load_to_lds(g0, g1, z4, z4, 0);
#endif
}
#endif

union Frag { v16bf v; v8bf h[2]; };

__device__ inline v8f zero8() {
  v8f z;
#pragma unroll
  for (int i = 0; i < 8; ++i) z[i] = 0.f;
  return z;
}

/* A fragment (16x32 bf16, ISA 7.12.2): lane%16 = M; slots 0..7 <-> K=k0+8*half+j,
   slots 8..15 <-> K=k0+16+8*half+j  => two contiguous 16B runs of the row. */
__device__ inline v16bf fragA_bf(const __bf16* row, int k0, int half) {
  Frag f;
  f.h[0] = *(const v8bf*)(row + k0 + 8 * half);
  f.h[1] = *(const v8bf*)(row + k0 + 16 + 8 * half);
  return f.v;
}

/* Same fragment built from fp32 activations with inline RNE converts. */
__device__ inline v16bf fragA_f32(const float* row, int k0, int half) {
  const float4* p0 = (const float4*)(row + k0 + 8 * half);
  const float4* p1 = (const float4*)(row + k0 + 16 + 8 * half);
  float4 a0 = p0[0], a1 = p0[1], b0 = p1[0], b1 = p1[1];
  Frag f;
  f.v[0] = (__bf16)a0.x; f.v[1] = (__bf16)a0.y; f.v[2]  = (__bf16)a0.z; f.v[3]  = (__bf16)a0.w;
  f.v[4] = (__bf16)a1.x; f.v[5] = (__bf16)a1.y; f.v[6]  = (__bf16)a1.z; f.v[7]  = (__bf16)a1.w;
  f.v[8] = (__bf16)b0.x; f.v[9] = (__bf16)b0.y; f.v[10] = (__bf16)b0.z; f.v[11] = (__bf16)b0.w;
  f.v[12] = (__bf16)b1.x; f.v[13] = (__bf16)b1.y; f.v[14] = (__bf16)b1.z; f.v[15] = (__bf16)b1.w;
  return f.v;
}

/* B fragment (32x16 bf16): lane%16 = N; slots 0..7 <-> K=k0+8*half+j,
   slots 8..15 <-> K=k0+16+8*half+j  => two contiguous 16B runs of row-major W. */
__device__ inline v16bf fragB(const __bf16* wrow, int k0, int half) {
  Frag f;
  f.h[0] = *(const v8bf*)(wrow + k0 + 8 * half);
  f.h[1] = *(const v8bf*)(wrow + k0 + 16 + 8 * half);
  return f.v;
}

__device__ inline v8f wmma_bf16(v16bf a, v16bf b, v8f c) {
  return __builtin_amdgcn_wmma_f32_16x16x32_bf16(false, a, false, b, (short)0, c,
                                                 false, false);
}

__device__ inline float sigmoidf_(float x) { return 1.f / (1.f + __expf(-x)); }

/* vectorized fp32 -> bf16 row staging into LDS (16B loads, 8B stores) */
__device__ inline void stage_bf16(const float* __restrict__ src, __bf16* dst,
                                  int n, int tid, int nthreads) {
  for (int i = tid * 4; i < n; i += nthreads * 4) {
    float4 fv = *(const float4*)(src + i);
    v4bf b;
    b[0] = (__bf16)fv.x; b[1] = (__bf16)fv.y; b[2] = (__bf16)fv.z; b[3] = (__bf16)fv.w;
    *(v4bf*)(dst + i) = b;
  }
}

/* ---------------- init ---------------- */
__global__ void k_init(float* hbufs, float* usage, float* csum, unsigned* cnt) {
  int i = blockIdx.x * 256 + threadIdx.x;
  if (i < 2 * B_ * H_) hbufs[i] = 0.f;
  if (i < E_) usage[i] = 0.f;
  if (i == E_) csum[0] = 0.f;
  if (i == E_ + 1) *cnt = 0u;
}

/* ---------------- weight conversion to bf16, natural [n][k] layout --------- */
__global__ void k_conv_wih(const float* __restrict__ W_ih, __bf16* __restrict__ out) {
  int i = blockIdx.x * 256 + threadIdx.x;     /* over 768*768 */
  if (i >= G3_ * D_) return;
  int n = i / D_, k = i % D_;
  out[i] = (__bf16)W_ih[(size_t)n * (D_ + E_) + k];
}
__global__ void k_conv_whh(const float* __restrict__ W_hh, __bf16* __restrict__ out) {
  int i = blockIdx.x * 256 + threadIdx.x;     /* over 768*256, same layout */
  if (i >= G3_ * H_) return;
  out[i] = (__bf16)W_hh[i];
}
__global__ void k_conv_whead(const float* __restrict__ Wc1, const float* __restrict__ Wr,
                             const float* __restrict__ Wk1, __bf16* __restrict__ out) {
  int i = blockIdx.x * 256 + threadIdx.x;     /* over 208*256 */
  if (i >= NH_ * H_) return;
  int n = i >> 8, k = i & 255;
  float v = 0.f;
  if (n < 128)       v = Wc1[(size_t)n * H_ + k];
  else if (n < 136)  v = Wr[(size_t)(n - 128) * H_ + k];
  else if (n >= 144) v = Wk1[(size_t)(n - 144) * H_ + k];
  out[i] = (__bf16)v;
}

/* -------- gi = x @ W_ih[:, :768].T + b_ih : LDS-free WMMA GEMM ------------- */
__global__ __launch_bounds__(256) void k_gemm_gi(const float* __restrict__ x,
                                                 const float* __restrict__ b_ih,
                                                 const __bf16* __restrict__ Wihb,
                                                 float* __restrict__ gi) {
  const int tid = threadIdx.x, lane = tid & 31, wv = tid >> 5;
  const int half = lane >> 4, ln = lane & 15;
  const int M0 = blockIdx.x * 32;
  const int mt = wv & 1;
  const int nt0 = (wv >> 1) * 12;             /* 12 N-tiles per wave, 48 total */
  const float* xrow = x + (size_t)(M0 + mt * 16 + ln) * D_;

  const __bf16* wrow[12];
#pragma unroll
  for (int j = 0; j < 12; ++j)
    wrow[j] = Wihb + (size_t)((nt0 + j) * 16 + ln) * D_;

  v8f acc[12];
#pragma unroll
  for (int j = 0; j < 12; ++j) acc[j] = zero8();

  for (int ks = 0; ks < 24; ++ks) {
    const int k0 = ks * 32;
    v16bf af = fragA_f32(xrow, k0, half);     /* fp32 x -> bf16 inline */
#pragma unroll
    for (int j = 0; j < 12; ++j)
      acc[j] = wmma_bf16(af, fragB(wrow[j], k0, half), acc[j]);
  }
#pragma unroll
  for (int j = 0; j < 12; ++j) {
    int ncol = (nt0 + j) * 16 + ln;
    float bias = b_ih[ncol];
#pragma unroll
    for (int r = 0; r < 8; ++r) {
      int row = M0 + mt * 16 + half * 8 + r;
      gi[(size_t)row * G3_ + ncol] = acc[j][r] + bias;
    }
  }
}

/* ---------------- GRU scan: 8 persistent WGPs, grid barrier per step -------- */
__global__ __launch_bounds__(128) void k_gru_scan(const float* __restrict__ gi,
                                                  const float* __restrict__ b_hh,
                                                  const __bf16* __restrict__ Whhb,
                                                  float* __restrict__ hbufs,
                                                  float* __restrict__ g_out,
                                                  float* __restrict__ out_hid,
                                                  unsigned* __restrict__ cnt) {
  __shared__ __bf16 WhhL[96][H_];  /* this block's 96 W_hh rows, natural [n][k] */
  __shared__ __bf16 hA[32][H_];
  __shared__ float  ghS[32][96];
  const int tid = threadIdx.x, lane = tid & 31, wv = tid >> 5;
  const int half = lane >> 4, ln = lane & 15;
  const int hc0 = blockIdx.x * 32;

  /* preload W_hh slice rows {gate*256 + hc0 + j} */
#if USE_TDM
  if (wv == 0) {                 /* uniform branch: exactly one wave issues TDM */
#pragma unroll
    for (int gate = 0; gate < 3; ++gate)
      tdm_load_2d(&WhhL[gate * 32][0],
                  Whhb + (size_t)(gate * H_ + hc0) * H_,
                  /*tile_rows=*/32, /*tile_cols=*/H_,
                  /*tensor_rows=*/G3_, /*row_stride=*/H_);
  }
  __builtin_amdgcn_s_wait_tensorcnt(0);
#else
  for (int i = tid; i < 96 * (H_ / 8); i += 128) {  /* 16B wide copies */
    int c = i / (H_ / 8), ck = i % (H_ / 8);
    int gate = c >> 5, j = c & 31;
    *(v8bf*)&WhhL[c][ck * 8] =
        *(const v8bf*)(Whhb + (size_t)(gate * H_ + hc0 + j) * H_ + ck * 8);
  }
#endif

  for (int t = 0; t < S_; ++t) {
    const float* hr = hbufs + (size_t)(t & 1) * (B_ * H_);
    float*       hw = hbufs + (size_t)((t & 1) ^ 1) * (B_ * H_);
    __syncthreads();
    stage_bf16(hr, &hA[0][0], 32 * H_, tid, 128);
    if (t + 1 < S_)
      __builtin_prefetch(gi + ((size_t)(tid >> 2) * S_ + (t + 1)) * G3_ + hc0, 0, 0);
    __syncthreads();

#pragma unroll
    for (int q = 0; q < 3; ++q) {
      int id = wv * 3 + q, mt = id & 1, nt = id >> 1;
      v8f acc = zero8();
#pragma unroll
      for (int ks = 0; ks < 8; ++ks) {
        v16bf af = fragA_bf(&hA[mt * 16 + ln][0], ks * 32, half);
        v16bf bf = fragB(&WhhL[nt * 16 + ln][0], ks * 32, half);
        acc = wmma_bf16(af, bf, acc);
      }
#pragma unroll
      for (int r = 0; r < 8; ++r)
        ghS[mt * 16 + half * 8 + r][nt * 16 + ln] = acc[r];
    }
    __syncthreads();

    for (int e = tid; e < 32 * 32; e += 128) {
      int b = e >> 5, j = e & 31, jg = hc0 + j;
      size_t row = (size_t)b * S_ + t;
      const float* gir = gi + row * G3_;
      float gr = ghS[b][j]      + b_hh[jg];
      float gz = ghS[b][32 + j] + b_hh[H_ + jg];
      float gn = ghS[b][64 + j] + b_hh[2 * H_ + jg];
      float r = sigmoidf_(gir[jg] + gr);
      float z = sigmoidf_(gir[H_ + jg] + gz);
      float n = tanhf(gir[2 * H_ + jg] + r * gn);
      float hp = hr[b * H_ + jg];
      float hn = (1.f - z) * n + z * hp;
      hw[b * H_ + jg] = hn;
      g_out[row * H_ + jg] = hn;
      if (t == S_ - 1) out_hid[b * H_ + jg] = hn;
    }

    /* device-scope grid barrier across the 8 persistent blocks */
    __syncthreads();
    if (tid == 0) {
      __threadfence();
      atomicAdd(cnt, 1u);
      unsigned target = (unsigned)(t + 1) * gridDim.x;
      while (__hip_atomic_load(cnt, __ATOMIC_ACQUIRE, __HIP_MEMORY_SCOPE_AGENT) < target)
        __builtin_amdgcn_s_sleep(1);
    }
    __syncthreads();
  }
}

/* ---------------- heads: fused WMMA GEMM + per-token epilogue --------------- */
__global__ __launch_bounds__(128) void k_heads(const float* __restrict__ g,
                                               const __bf16* __restrict__ Whb,
                                               const float* __restrict__ bc1,
                                               const float* __restrict__ Wc2,
                                               const float* __restrict__ bc2,
                                               const float* __restrict__ br,
                                               const float* __restrict__ bk1,
                                               const float* __restrict__ Wk2,
                                               const float* __restrict__ bk2,
                                               float* __restrict__ out,
                                               float* __restrict__ usage,
                                               float* __restrict__ csum) {
  __shared__ __bf16 WB[NH_][H_];   /* natural [n][k] */
  __shared__ __bf16 gA[32][H_];
  __shared__ float  outS[32][NH_];
  __shared__ float  bu[E_];
  __shared__ float  bcs;
  const int tid = threadIdx.x, lane = tid & 31, wv = tid >> 5;
  const int half = lane >> 4, ln = lane & 15;
  const int M0 = blockIdx.x * 32;

  if (tid < E_) bu[tid] = 0.f;
  if (tid == E_) bcs = 0.f;
#if USE_TDM
  if (wv == 0)
    tdm_load_2d(&WB[0][0], Whb, /*tile_rows=*/NH_, /*tile_cols=*/H_,
                /*tensor_rows=*/NH_, /*row_stride=*/H_);
  __builtin_amdgcn_s_wait_tensorcnt(0);
#else
  {
    const uint4* src = (const uint4*)Whb;               /* contiguous copy */
    uint4* dst = (uint4*)&WB[0][0];
    for (int i = tid; i < (NH_ * H_ * 2) / 16; i += 128) dst[i] = src[i];
  }
#endif
  stage_bf16(g + (size_t)M0 * H_, &gA[0][0], 32 * H_, tid, 128);
  __syncthreads();

  for (int id = wv; id < 26; id += 4) { /* 2 M-tiles x 13 N-tiles */
    int mt = id & 1, nt = id >> 1;
    v8f acc = zero8();
#pragma unroll
    for (int ks = 0; ks < 8; ++ks) {
      v16bf af = fragA_bf(&gA[mt * 16 + ln][0], ks * 32, half);
      v16bf bf = fragB(&WB[nt * 16 + ln][0], ks * 32, half);
      acc = wmma_bf16(af, bf, acc);
    }
    int n = nt * 16 + ln;
    float bias; bool relu;
    if (n < 128)      { bias = bc1[n];       relu = true;  }
    else if (n < 136) { bias = br[n - 128];  relu = false; }
    else if (n < 144) { bias = 0.f;          relu = false; }
    else              { bias = bk1[n - 144]; relu = true;  }
#pragma unroll
    for (int r = 0; r < 8; ++r) {
      float v = acc[r] + bias;
      if (relu) v = fmaxf(v, 0.f);
      outS[mt * 16 + half * 8 + r][n] = v;
    }
  }
  __syncthreads();

  if (tid < 32) {
    const int tok = M0 + tid;
    float c1d = bc2[0];
    for (int k = 0; k < 128; ++k) c1d += outS[tid][k] * Wc2[k];
    float cplx = sigmoidf_(c1d);
    float kd = bk2[0];
    for (int k = 0; k < 64; ++k) kd += outS[tid][144 + k] * Wk2[k];
    float tks = sigmoidf_(kd);

    float lg[E_], mx = -1e30f;
    for (int e = 0; e < E_; ++e) { lg[e] = outS[tid][128 + e]; mx = fmaxf(mx, lg[e]); }
    float ssum = 0.f, rw[E_];
    for (int e = 0; e < E_; ++e) { rw[e] = __expf(lg[e] - mx); ssum += rw[e]; }
    float inv = 1.f / ssum;
    for (int e = 0; e < E_; ++e) {
      rw[e] *= inv;
      out[OUT_RW + (size_t)tok * E_ + e] = rw[e];
    }
    float comb = 0.7f * cplx + 0.3f * tks;
    int dynk = (int)rintf(1.f + comb * 3.f); /* round-half-even == jnp.round */

    float tw[4]; int ti[4];
    float tmp[E_];
    for (int e = 0; e < E_; ++e) tmp[e] = rw[e];
    for (int j = 0; j < 4; ++j) {
      int am = 0; float bv = tmp[0];
      for (int e = 1; e < E_; ++e) if (tmp[e] > bv) { bv = tmp[e]; am = e; }
      tw[j] = bv; ti[j] = am; tmp[am] = -1.f;
    }
    float ms = 0.f;
    for (int j = 0; j < 4; ++j) if (j < dynk) ms += tw[j];
    float den = (ms > 0.f) ? ms : 1.f;
    for (int j = 0; j < 4; ++j) {
      float ewv = (j < dynk) ? tw[j] / den : 0.f;
      out[OUT_EW  + (size_t)tok * 4 + j] = ewv;
      out[OUT_IDX + (size_t)tok * 4 + j] = (float)ti[j];
    }
    out[OUT_CPLX + tok] = cplx;

    for (int e = 0; e < E_; ++e) atomicAdd(&bu[e], rw[e]);
    atomicAdd(&bcs, cplx * cplx);
  }
  __syncthreads();
  if (tid < E_) atomicAdd(&usage[tid], bu[tid]);
  if (tid == E_) atomicAdd(csum, bcs);
}

/* ---------------- losses ---------------- */
__global__ void k_finalize(const float* __restrict__ usage, const float* __restrict__ csum,
                           float* __restrict__ out) {
  if (threadIdx.x == 0 && blockIdx.x == 0) {
    float lb = 0.f;
    for (int e = 0; e < E_; ++e) {
      float d = usage[e] / (float)BS_ - 1.f / (float)E_;
      lb += d * d;
    }
    out[OUT_LB]   = (lb / (float)E_) * 0.01f;
    out[OUT_CREG] = (csum[0] / (float)BS_) * 0.001f;
  }
}

extern "C" void kernel_launch(void* const* d_in, const int* in_sizes, int n_in,
                              void* d_out, int out_size, void* d_ws, size_t ws_size,
                              hipStream_t stream) {
  (void)in_sizes; (void)n_in; (void)out_size; (void)ws_size;
  const float* x    = (const float*)d_in[0];
  const float* W_ih = (const float*)d_in[2];
  const float* W_hh = (const float*)d_in[3];
  const float* b_ih = (const float*)d_in[4];
  const float* b_hh = (const float*)d_in[5];
  const float* Wc1  = (const float*)d_in[6];
  const float* bc1  = (const float*)d_in[7];
  const float* Wc2  = (const float*)d_in[8];
  const float* bc2  = (const float*)d_in[9];
  const float* Wr   = (const float*)d_in[10];
  const float* br   = (const float*)d_in[11];
  const float* Wk1  = (const float*)d_in[12];
  const float* bk1  = (const float*)d_in[13];
  const float* Wk2  = (const float*)d_in[14];
  const float* bk2  = (const float*)d_in[15];

  float* out = (float*)d_out;
  char*  ws  = (char*)d_ws;
  float*    gi     = (float*)(ws + WS_GI);
  float*    gbuf   = (float*)(ws + WS_G);
  float*    hbufs  = (float*)(ws + WS_H);
  __bf16*   wihb   = (__bf16*)(ws + WS_WIH);
  __bf16*   whhb   = (__bf16*)(ws + WS_WHH);
  __bf16*   wheadb = (__bf16*)(ws + WS_WHEAD);
  float*    usage  = (float*)(ws + WS_USAGE);
  float*    csum   = (float*)(ws + WS_CSUM);
  unsigned* cnt    = (unsigned*)(ws + WS_CNT);

  k_init<<<64, 256, 0, stream>>>(hbufs, usage, csum, cnt);
  k_conv_wih<<<(G3_ * D_ + 255) / 256, 256, 0, stream>>>(W_ih, wihb);
  k_conv_whh<<<(G3_ * H_ + 255) / 256, 256, 0, stream>>>(W_hh, whhb);
  k_conv_whead<<<(NH_ * H_ + 255) / 256, 256, 0, stream>>>(Wc1, Wr, Wk1, wheadb);
  k_gemm_gi<<<BS_ / 32, 256, 0, stream>>>(x, b_ih, wihb, gi);
  k_gru_scan<<<8, 128, 0, stream>>>(gi, b_hh, whhb, hbufs, gbuf, out + OUT_HID, cnt);
  k_heads<<<BS_ / 32, 128, 0, stream>>>(gbuf, wheadb, bc1, Wc2, bc2, br, bk1, Wk2, bk2,
                                        out, usage, csum);
  k_finalize<<<1, 32, 0, stream>>>(usage, csum, out);
}
